// CEBlock_18597208392036
// MI455X (gfx1250) — matrix operand order repaired and checked by reading, not compiled, not run on hardware
//
#include <hip/hip_runtime.h>

// ---------------------------------------------------------------------------
// CEBlock for MI455X (gfx1250, wave32, WMMA). All GEMMs run on
// v_wmma_f32_16x16x32_bf16 with fp32 accumulation; residuals/softmax in fp32.
// Global->LDS staging uses gfx1250 async-to-LDS DMA (ASYNCcnt) with
// double-buffered LDS so the next K-slab streams in under the WMMAs.
// ---------------------------------------------------------------------------

#define Bb   32
#define NT   64
#define NS   256
#define NN   320      // NT + NS
#define NPS  256
#define Dd   768
#define Hh   12
#define DHd  64
#define LK   180      // kept search tokens
#define NREM 76       // NS - LK
#define NF   244      // NT + LK

typedef __attribute__((ext_vector_type(16))) __bf16 v16bf;
typedef __attribute__((ext_vector_type(8)))  float  v8f;

__device__ __forceinline__ unsigned short f2bf(float f) {
  unsigned int u = __float_as_uint(f);
  u += 0x7fffu + ((u >> 16) & 1u);            // round-to-nearest-even
  return (unsigned short)(u >> 16);
}

// Low 32 bits of a generic LDS pointer are the LDS byte offset (flat aperture
// places the aperture tag entirely in addr[63:32]); async-to-LDS VDST wants
// exactly that 32-bit LDS address.
__device__ __forceinline__ unsigned lds_addr32(const void* p) {
  return (unsigned)(unsigned long long)p;
}

// ---------------------------------------------------------------------------
// Generic bf16 WMMA GEMM:  C[M,Nc] = epilogue( A[M,K] * Bt[Nc,K]^T )
//   A   : bf16 row-major, leading dim lda, per-z stride sAz (elements)
//   Bt  : bf16, row n holds output-column n over K (i.e. W^T), ld ldb
//   EPI 0: f32 out = res + acc + bias          (residual GEMMs)
//   EPI 1: f32 out = acc * scale               (attention logits)
//   EPI 2: bf16 out = acc                      (projections, AV)
//   EPI 3: bf16 out = gelu(acc + bias)         (MLP up)
//   Output/residual row remap: crow = (m/rpb)*brs + rofs + m%rpb
// Block: 256 thr = 8 waves; tile 128x64x32; each wave does 2x2 WMMA tiles.
// A-rows are clamped (not zero-filled): OOB rows only feed accumulator rows
// that the epilogue masks, so the hot loop stays divergence-free (EXEC all-1s,
// as WMMA requires).
// ---------------------------------------------------------------------------
#define BM  128
#define BN  64
#define BK  32
#define BKP 40   // +8 bf16 pad: row stride 80B (16B aligned, conflict-mitigating)

template <int EPI>
__global__ __launch_bounds__(256) void gemm_wmma(
    const unsigned short* __restrict__ A, int lda, long sAz,
    const unsigned short* __restrict__ Bt, int ldb, long sBz,
    const float* __restrict__ bias, const float* __restrict__ res,
    void* __restrict__ Cout, int ldc, long sCz,
    int M, int Nc, int K, float scale, int rpb, int brs, int rofs)
{
  __shared__ __align__(16) unsigned short Ash[2][BM * BKP];
  __shared__ __align__(16) unsigned short Bsh[2][BN * BKP];

  const int z = blockIdx.z;
  A  += (long)z * sAz;
  Bt += (long)z * sBz;
  const long cbase = (long)z * sCz;

  const int bm0 = blockIdx.x * BM;
  const int bn0 = blockIdx.y * BN;
  const int tid  = threadIdx.x;
  const int lane = tid & 31;
  const int wid  = tid >> 5;
  const int wm   = (wid & 3) * 32;   // wave tile M offset (4 waves over M)
  const int wn   = (wid >> 2) * 32;  // wave tile N offset (2 waves over N)
  const int half = lane >> 4;
  const int l15  = lane & 15;

  v8f acc[2][2] = {};
  union F16 { v16bf v; uint4 q[2]; };

  const int arow = tid >> 1, akc = (tid & 1) << 4;      // A: 2 thr/row, 16 elems
  const int brow = tid >> 2, bkc = (tid & 3) << 3;      // B: 4 thr/row, 8 elems
  long gra = (long)bm0 + arow;
  if (gra > M - 1) gra = M - 1;                          // clamp, no divergence

  const unsigned ldsA0 = lds_addr32(&Ash[0][0]) + (unsigned)(arow * BKP + akc) * 2u;
  const unsigned ldsB0 = lds_addr32(&Bsh[0][0]) + (unsigned)(brow * BKP + bkc) * 2u;
  const unsigned short* gA = A + gra * (long)lda + akc;
  const unsigned short* gB = Bt + (long)(bn0 + brow) * ldb + bkc;

  // Issue this thread's share of tile kt into LDS buffer `buf` (fire & forget,
  // tracked by ASYNCcnt). A: 2x16B, B: 1x16B per thread.
  auto issue_tile = [&](int kt, int buf) {
    const unsigned la = ldsA0 + (unsigned)buf * (unsigned)(BM * BKP * 2);
    const unsigned lb = ldsB0 + (unsigned)buf * (unsigned)(BN * BKP * 2);
    unsigned long long ga = (unsigned long long)(const void*)(gA + kt * BK);
    unsigned long long gb = (unsigned long long)(const void*)(gB + kt * BK);
    asm volatile("global_load_async_to_lds_b128 %0, %1, off"
                 :: "v"(la), "v"(ga) : "memory");
    asm volatile("global_load_async_to_lds_b128 %0, %1, off offset:16"
                 :: "v"(la + 16u), "v"(ga) : "memory");
    asm volatile("global_load_async_to_lds_b128 %0, %1, off"
                 :: "v"(lb), "v"(gb) : "memory");
  };

  const int nk = K / BK;
  issue_tile(0, 0);

  for (int kt = 0; kt < nk; ++kt) {
    asm volatile("s_wait_asynccnt 0x0" ::: "memory");   // own DMAs landed
    __syncthreads();                                    // everyone's landed
    if (kt + 1 < nk) issue_tile(kt + 1, (kt + 1) & 1);  // prefetch next slab

    const unsigned short* As = &Ash[kt & 1][0];
    const unsigned short* Bs = &Bsh[kt & 1][0];

    // ---- fragments per ISA layout ----
    F16 fa[2], fb[2];
#pragma unroll
    for (int mi = 0; mi < 2; mi++) {
      const unsigned short* ap = As + (wm + mi * 16 + l15) * BKP + 8 * half;
      fa[mi].q[0] = *(const uint4*)(ap);        // e0..7  : K = 8*half + e
      fa[mi].q[1] = *(const uint4*)(ap + 16);   // e8..15 : K = 16 + 8*half + (e-8)
    }
#pragma unroll
    for (int ni = 0; ni < 2; ni++) {
      const unsigned short* bp = Bs + (wn + ni * 16 + l15) * BKP + 8 * half;
      fb[ni].q[0] = *(const uint4*)(bp);        // e0..7  : K = 8*half + e
      fb[ni].q[1] = *(const uint4*)(bp + 16);
    }

#pragma unroll
    for (int mi = 0; mi < 2; mi++)
#pragma unroll
      for (int ni = 0; ni < 2; ni++)
        acc[mi][ni] = __builtin_amdgcn_wmma_f32_16x16x32_bf16(
            false, fa[mi].v, false, fb[ni].v, (short)0, acc[mi][ni], false, false);

    __syncthreads();   // all waves done reading buf (kt&1) before it is reused
  }

  // ---- epilogue: C element r of lane l -> (M = tile + 8*half + r, N = l&15)
#pragma unroll
  for (int mi = 0; mi < 2; mi++)
#pragma unroll
    for (int ni = 0; ni < 2; ni++) {
      v8f a = acc[mi][ni];
#pragma unroll
      for (int r = 0; r < 8; r++) {
        int gm = bm0 + wm + mi * 16 + 8 * half + r;
        int gn = bn0 + wn + ni * 16 + l15;
        if (gm < M) {
          long cr = (long)(gm / rpb) * brs + rofs + (gm % rpb);
          long o  = cbase + cr * (long)ldc + gn;
          float v = a[r];
          if (EPI == 1) {
            ((float*)Cout)[o] = v * scale;
          } else if (EPI == 0) {
            ((float*)Cout)[o] = res[o] + v + bias[gn];
          } else if (EPI == 2) {
            ((unsigned short*)Cout)[o] = f2bf(v);
          } else {
            float t = v + bias[gn];
            ((unsigned short*)Cout)[o] =
                f2bf(0.5f * t * (1.0f + erff(t * 0.70710678118f)));
          }
        }
      }
    }
}

// ---------------------------------------------------------------------------
// LayerNorm (D = 768) f32 -> bf16 with row remap (handles the x2 s-row view).
// ---------------------------------------------------------------------------
__global__ __launch_bounds__(256) void ln_bf16_kernel(
    const float* __restrict__ in, const float* __restrict__ g,
    const float* __restrict__ be, unsigned short* __restrict__ out,
    int rpb, int brs, int rofs)
{
  const int r = blockIdx.x, tid = threadIdx.x;
  const long irow = ((long)(r / rpb) * brs + rofs + (r % rpb)) * Dd;
  float vals[3], s = 0.f, s2 = 0.f;
#pragma unroll
  for (int i = 0; i < 3; i++) {
    float v = in[irow + tid + i * 256];
    vals[i] = v; s += v; s2 += v * v;
  }
  for (int off = 16; off > 0; off >>= 1) {
    s  += __shfl_down(s, off);
    s2 += __shfl_down(s2, off);
  }
  __shared__ float shs[8], shs2[8], mb[2];
  if ((tid & 31) == 0) { shs[tid >> 5] = s; shs2[tid >> 5] = s2; }
  __syncthreads();
  if (tid == 0) {
    float S = 0.f, S2 = 0.f;
    for (int i = 0; i < 8; i++) { S += shs[i]; S2 += shs2[i]; }
    float m = S / (float)Dd;
    mb[0] = m;
    mb[1] = rsqrtf(S2 / (float)Dd - m * m + 1e-5f);
  }
  __syncthreads();
  float m = mb[0], rs = mb[1];
#pragma unroll
  for (int i = 0; i < 3; i++) {
    int c = tid + i * 256;
    out[(long)r * Dd + c] = f2bf((vals[i] - m) * rs * g[c] + be[c]);
  }
}

// ---------------------------------------------------------------------------
// Row softmax: f32 in/out (optional) + bf16 copy for the AV WMMA GEMM.
// ---------------------------------------------------------------------------
__global__ __launch_bounds__(256) void softmax_kernel(
    float* __restrict__ io, unsigned short* __restrict__ obf, int L, int wb)
{
  const long base = (long)blockIdx.x * L;
  const int tid = threadIdx.x, lane = tid & 31, w = tid >> 5;
  __shared__ float sw[8];

  float mx = -3.4e38f;
  for (int i = tid; i < L; i += 256) mx = fmaxf(mx, io[base + i]);
  for (int off = 16; off > 0; off >>= 1) mx = fmaxf(mx, __shfl_xor(mx, off));
  if (lane == 0) sw[w] = mx;
  __syncthreads();
  float gm = sw[0];
#pragma unroll
  for (int i = 1; i < 8; i++) gm = fmaxf(gm, sw[i]);
  __syncthreads();

  float s = 0.f;
  for (int i = tid; i < L; i += 256) s += expf(io[base + i] - gm);
  for (int off = 16; off > 0; off >>= 1) s += __shfl_xor(s, off);
  if (lane == 0) sw[w] = s;
  __syncthreads();
  float tot = 0.f;
#pragma unroll
  for (int i = 0; i < 8; i++) tot += sw[i];
  float inv = 1.f / tot;

  for (int i = tid; i < L; i += 256) {
    float e = expf(io[base + i] - gm) * inv;
    if (wb) io[base + i] = e;
    obf[base + i] = f2bf(e);
  }
}

// attn_t[b][j] = mean over (h, q<NT) of attn[b][h][q][NT+j]
__global__ __launch_bounds__(256) void attn_mean_kernel(
    const float* __restrict__ attn, float* __restrict__ attn_t)
{
  const int b = blockIdx.x, j = threadIdx.x;
  const float* base = attn + (long)b * Hh * NN * NN;
  float s = 0.f;
  for (int h = 0; h < Hh; h++)
    for (int q = 0; q < NT; q++)
      s += base[((long)h * NN + q) * NN + NT + j];
  attn_t[b * NS + j] = s * (1.0f / (Hh * NT));
}

// Stable descending rank (== stable argsort of -attn_t); emit indices.
__global__ __launch_bounds__(256) void topk_kernel(
    const float* __restrict__ attn_t, const int* __restrict__ gis,
    float* __restrict__ keep_out, float* __restrict__ rem_out,
    int* __restrict__ topk_idx)
{
  const int b = blockIdx.x, j = threadIdx.x;
  __shared__ float vals[NS];
  float vj = attn_t[b * NS + j];
  vals[j] = vj;
  __syncthreads();
  int rank = 0;
  for (int i = 0; i < NS; i++) {
    float vi = vals[i];
    rank += (vi > vj) || (vi == vj && i < j);
  }
  int gi = gis[b * NS + j];
  if (rank < LK) {
    keep_out[b * LK + rank] = (float)gi;
    topk_idx[b * LK + rank] = j;
  } else {
    rem_out[b * NREM + (rank - LK)] = (float)gi;
  }
}

// x2 (B,244,768) = concat(x1[:, :NT], gather(x1[:, NT:], topk))
__global__ __launch_bounds__(256) void gather_kernel(
    const float* __restrict__ x1, const int* __restrict__ topk_idx,
    float* __restrict__ x2)
{
  long i = (long)blockIdx.x * 256 + threadIdx.x;
  const long total = (long)Bb * NF * Dd;
  if (i >= total) return;
  int d = (int)(i % Dd);
  long t = i / Dd;
  int row = (int)(t % NF), b = (int)(t / NF);
  long src;
  if (row < NT) src = ((long)b * NN + row) * Dd + d;
  else          src = ((long)b * NN + NT + topk_idx[b * LK + row - NT]) * Dd + d;
  x2[i] = x1[src];
}

__global__ __launch_bounds__(256) void git_kernel(const int* __restrict__ g,
                                                  float* __restrict__ o) {
  int i = blockIdx.x * 256 + threadIdx.x;
  if (i < Bb * NT) o[i] = (float)g[i];
}

// W (K,N) f32 row-major  ->  Wt (N,K) bf16 (so B-fragments are contiguous)
__global__ __launch_bounds__(256) void wconv_kernel(
    const float* __restrict__ W, unsigned short* __restrict__ Wt, int K, int Nc)
{
  long i = (long)blockIdx.x * 256 + threadIdx.x;
  if (i >= (long)K * Nc) return;
  int n = (int)(i % Nc);
  long k = i / Nc;
  Wt[(long)n * K + k] = f2bf(W[i]);
}

__global__ __launch_bounds__(256) void cvtbf_kernel(
    const float* __restrict__ in, unsigned short* __restrict__ out, long n) {
  long i = (long)blockIdx.x * 256 + threadIdx.x;
  if (i < n) out[i] = f2bf(in[i]);
}

// (B,rows,stride)[colOff + h*64 + d] -> (B,H,rows,64)
__global__ __launch_bounds__(256) void to_heads_kernel(
    const unsigned short* __restrict__ src, int srcStride, int colOff,
    unsigned short* __restrict__ dst, int rows)
{
  long i = (long)blockIdx.x * 256 + threadIdx.x;
  const long total = (long)Bb * Hh * rows * DHd;
  if (i >= total) return;
  int d = (int)(i % DHd);
  long r = i / DHd;
  int t = (int)(r % rows); r /= rows;
  int h = (int)(r % Hh), b = (int)(r / Hh);
  dst[i] = src[((long)b * rows + t) * srcStride + colOff + h * DHd + d];
}

// -> (B,H,64,rows)  (transposed V so AV B-fragments are contiguous over K)
__global__ __launch_bounds__(256) void to_headsT_kernel(
    const unsigned short* __restrict__ src, int srcStride, int colOff,
    unsigned short* __restrict__ dst, int rows)
{
  long i = (long)blockIdx.x * 256 + threadIdx.x;
  const long total = (long)Bb * Hh * rows * DHd;
  if (i >= total) return;
  int t = (int)(i % rows);
  long r = i / rows;
  int d = (int)(r % DHd); r /= DHd;
  int h = (int)(r % Hh), b = (int)(r / Hh);
  dst[i] = src[((long)b * rows + t) * srcStride + colOff + h * DHd + d];
}

// (B,H,rows,64) -> (B,rows,768)
__global__ __launch_bounds__(256) void merge_heads_kernel(
    const unsigned short* __restrict__ src, unsigned short* __restrict__ dst,
    int rows)
{
  long i = (long)blockIdx.x * 256 + threadIdx.x;
  const long total = (long)Bb * Hh * rows * DHd;
  if (i >= total) return;
  int d = (int)(i % DHd);
  long r = i / DHd;
  int t = (int)(r % rows); r /= rows;
  int h = (int)(r % Hh), b = (int)(r / Hh);
  dst[((long)b * rows + t) * Dd + h * DHd + d] = src[i];
}

// ---------------------------------------------------------------------------
extern "C" void kernel_launch(void* const* d_in, const int* in_sizes, int n_in,
                              void* d_out, int out_size, void* d_ws,
                              size_t ws_size, hipStream_t stream)
{
  const float* x     = (const float*)d_in[0];
  const float* ps    = (const float*)d_in[1];
  const int*   git   = (const int*)d_in[2];
  const int*   gis   = (const int*)d_in[3];
  const float* g1c   = (const float*)d_in[4];
  const float* b1c   = (const float*)d_in[5];
  const float* Wqkv  = (const float*)d_in[6];
  const float* Wproj = (const float*)d_in[7];
  const float* bproj = (const float*)d_in[8];
  const float* gtc   = (const float*)d_in[9];
  const float* btc   = (const float*)d_in[10];
  const float* Wq    = (const float*)d_in[11];
  const float* Wk    = (const float*)d_in[12];
  const float* Wv    = (const float*)d_in[13];
  const float* Wtp   = (const float*)d_in[14];
  const float* btp   = (const float*)d_in[15];
  const float* g2c   = (const float*)d_in[16];
  const float* b2c   = (const float*)d_in[17];
  const float* W1    = (const float*)d_in[18];
  const float* b1m   = (const float*)d_in[19];
  const float* W2    = (const float*)d_in[20];
  const float* b2m   = (const float*)d_in[21];

  float* out_x    = (float*)d_out;
  float* out_git  = out_x + (long)Bb * NF * Dd;
  float* out_keep = out_git + Bb * NT;
  float* out_rem  = out_keep + Bb * LK;
  float* out_attn = out_rem + Bb * NREM;

  char* wp = (char*)d_ws;
  auto alloc = [&](size_t bytes) -> void* {
    void* p = (void*)wp;
    wp += (bytes + 255) & ~(size_t)255;
    return p;
  };
  auto bf = [&](long elems) { return (unsigned short*)alloc((size_t)elems * 2); };
  auto f32 = [&](long elems) { return (float*)alloc((size_t)elems * 4); };

  unsigned short* WqkvT = bf((long)3 * Dd * Dd);
  unsigned short* WprojT = bf((long)Dd * Dd);
  unsigned short* WqT = bf((long)Dd * Dd);
  unsigned short* WkT = bf((long)Dd * Dd);
  unsigned short* WvT = bf((long)Dd * Dd);
  unsigned short* WtpT = bf((long)Dd * Dd);
  unsigned short* W1T = bf((long)4 * Dd * Dd);
  unsigned short* W2T = bf((long)4 * Dd * Dd);

  unsigned short* hbf   = bf((long)Bb * NN * Dd);
  unsigned short* qkvb  = bf((long)Bb * NN * 3 * Dd);
  unsigned short* qh    = bf((long)Bb * Hh * NN * DHd);
  unsigned short* kh    = bf((long)Bb * Hh * NN * DHd);
  unsigned short* vTh   = bf((long)Bb * Hh * DHd * NN);
  unsigned short* attnb = bf((long)Bb * Hh * NN * NN);
  unsigned short* xah   = bf((long)Bb * Hh * NN * DHd);
  unsigned short* xab   = bf((long)Bb * NN * Dd);
  float* x1             = f32((long)Bb * NN * Dd);
  float* attn_t         = f32((long)Bb * NS);
  int* topk_idx         = (int*)alloc((size_t)Bb * LK * 4);
  float* x2             = f32((long)Bb * NF * Dd);
  unsigned short* snb   = bf((long)Bb * LK * Dd);
  unsigned short* psb   = bf((long)Bb * NPS * Dd);
  unsigned short* q2b   = bf((long)Bb * LK * Dd);
  unsigned short* kb    = bf((long)Bb * NPS * Dd);
  unsigned short* vb    = bf((long)Bb * NPS * Dd);
  unsigned short* q2h   = bf((long)Bb * Hh * LK * DHd);
  unsigned short* k2h   = bf((long)Bb * Hh * NPS * DHd);
  unsigned short* v2Th  = bf((long)Bb * Hh * DHd * NPS);
  float* s2f            = f32((long)Bb * Hh * LK * NPS);
  unsigned short* a2b   = bf((long)Bb * Hh * LK * NPS);
  unsigned short* sah   = bf((long)Bb * Hh * LK * DHd);
  unsigned short* sab   = bf((long)Bb * LK * Dd);
  unsigned short* h2b   = bf((long)Bb * NF * Dd);
  unsigned short* m1    = bf((long)Bb * NF * 4 * Dd);

  auto gemm = [&](int epi, const void* A, int lda, long sAz, const void* Bt,
                  int ldb, long sBz, const float* bias, const float* res,
                  void* C, int ldc, long sCz, int M, int Nc, int K, int Z,
                  float scale, int rpb, int brs, int rofs) {
    dim3 g((M + BM - 1) / BM, Nc / BN, Z), blk(256);
    const unsigned short* Au = (const unsigned short*)A;
    const unsigned short* Bu = (const unsigned short*)Bt;
    switch (epi) {
      case 0: gemm_wmma<0><<<g, blk, 0, stream>>>(Au, lda, sAz, Bu, ldb, sBz, bias, res, C, ldc, sCz, M, Nc, K, scale, rpb, brs, rofs); break;
      case 1: gemm_wmma<1><<<g, blk, 0, stream>>>(Au, lda, sAz, Bu, ldb, sBz, bias, res, C, ldc, sCz, M, Nc, K, scale, rpb, brs, rofs); break;
      case 2: gemm_wmma<2><<<g, blk, 0, stream>>>(Au, lda, sAz, Bu, ldb, sBz, bias, res, C, ldc, sCz, M, Nc, K, scale, rpb, brs, rofs); break;
      default: gemm_wmma<3><<<g, blk, 0, stream>>>(Au, lda, sAz, Bu, ldb, sBz, bias, res, C, ldc, sCz, M, Nc, K, scale, rpb, brs, rofs); break;
    }
  };
  auto g1d = [](long n) { return dim3((unsigned)((n + 255) / 256)); };

  // ---- weights -> bf16 transposed ----
  wconv_kernel<<<g1d((long)Dd * 3 * Dd), 256, 0, stream>>>(Wqkv, WqkvT, Dd, 3 * Dd);
  wconv_kernel<<<g1d((long)Dd * Dd), 256, 0, stream>>>(Wproj, WprojT, Dd, Dd);
  wconv_kernel<<<g1d((long)Dd * Dd), 256, 0, stream>>>(Wq, WqT, Dd, Dd);
  wconv_kernel<<<g1d((long)Dd * Dd), 256, 0, stream>>>(Wk, WkT, Dd, Dd);
  wconv_kernel<<<g1d((long)Dd * Dd), 256, 0, stream>>>(Wv, WvT, Dd, Dd);
  wconv_kernel<<<g1d((long)Dd * Dd), 256, 0, stream>>>(Wtp, WtpT, Dd, Dd);
  wconv_kernel<<<g1d((long)Dd * 4 * Dd), 256, 0, stream>>>(W1, W1T, Dd, 4 * Dd);
  wconv_kernel<<<g1d((long)4 * Dd * Dd), 256, 0, stream>>>(W2, W2T, 4 * Dd, Dd);

  // ---- stage 1: LN -> QKV -> attention -> proj residual ----
  ln_bf16_kernel<<<Bb * NN, 256, 0, stream>>>(x, g1c, b1c, hbf, NN, NN, 0);
  gemm(2, hbf, Dd, 0, WqkvT, Dd, 0, nullptr, nullptr, qkvb, 3 * Dd, 0,
       Bb * NN, 3 * Dd, Dd, 1, 0.f, Bb * NN, Bb * NN, 0);

  const long hc1 = (long)Bb * Hh * NN * DHd;
  to_heads_kernel<<<g1d(hc1), 256, 0, stream>>>(qkvb, 3 * Dd, 0, qh, NN);
  to_heads_kernel<<<g1d(hc1), 256, 0, stream>>>(qkvb, 3 * Dd, Dd, kh, NN);
  to_headsT_kernel<<<g1d(hc1), 256, 0, stream>>>(qkvb, 3 * Dd, 2 * Dd, vTh, NN);

  gemm(1, qh, DHd, (long)NN * DHd, kh, DHd, (long)NN * DHd, nullptr, nullptr,
       out_attn, NN, (long)NN * NN, NN, NN, DHd, Bb * Hh, 0.125f, NN, NN, 0);
  softmax_kernel<<<Bb * Hh * NN, 256, 0, stream>>>(out_attn, attnb, NN, 1);
  gemm(2, attnb, NN, (long)NN * NN, vTh, NN, (long)DHd * NN, nullptr, nullptr,
       xah, DHd, (long)NN * DHd, NN, DHd, NN, Bb * Hh, 0.f, NN, NN, 0);
  merge_heads_kernel<<<g1d(hc1), 256, 0, stream>>>(xah, xab, NN);
  gemm(0, xab, Dd, 0, WprojT, Dd, 0, bproj, x, x1, Dd, 0,
       Bb * NN, Dd, Dd, 1, 0.f, Bb * NN, Bb * NN, 0);

  // ---- pruning ----
  attn_mean_kernel<<<Bb, 256, 0, stream>>>(out_attn, attn_t);
  topk_kernel<<<Bb, 256, 0, stream>>>(attn_t, gis, out_keep, out_rem, topk_idx);
  gather_kernel<<<g1d((long)Bb * NF * Dd), 256, 0, stream>>>(x1, topk_idx, x2);
  git_kernel<<<g1d(Bb * NT), 256, 0, stream>>>(git, out_git);

  // ---- stage 2: cross attention on kept tokens vs ps ----
  ln_bf16_kernel<<<Bb * LK, 256, 0, stream>>>(x2, gtc, btc, snb, LK, NF, NT);
  cvtbf_kernel<<<g1d((long)Bb * NPS * Dd), 256, 0, stream>>>(ps, psb,
                                                             (long)Bb * NPS * Dd);
  gemm(2, snb, Dd, 0, WqT, Dd, 0, nullptr, nullptr, q2b, Dd, 0,
       Bb * LK, Dd, Dd, 1, 0.f, Bb * LK, Bb * LK, 0);
  gemm(2, psb, Dd, 0, WkT, Dd, 0, nullptr, nullptr, kb, Dd, 0,
       Bb * NPS, Dd, Dd, 1, 0.f, Bb * NPS, Bb * NPS, 0);
  gemm(2, psb, Dd, 0, WvT, Dd, 0, nullptr, nullptr, vb, Dd, 0,
       Bb * NPS, Dd, Dd, 1, 0.f, Bb * NPS, Bb * NPS, 0);

  const long hc2 = (long)Bb * Hh * LK * DHd;
  const long hc3 = (long)Bb * Hh * NPS * DHd;
  to_heads_kernel<<<g1d(hc2), 256, 0, stream>>>(q2b, Dd, 0, q2h, LK);
  to_heads_kernel<<<g1d(hc3), 256, 0, stream>>>(kb, Dd, 0, k2h, NPS);
  to_headsT_kernel<<<g1d(hc3), 256, 0, stream>>>(vb, Dd, 0, v2Th, NPS);

  gemm(1, q2h, DHd, (long)LK * DHd, k2h, DHd, (long)NPS * DHd, nullptr, nullptr,
       s2f, NPS, (long)LK * NPS, LK, NPS, DHd, Bb * Hh, 0.125f, LK, LK, 0);
  softmax_kernel<<<Bb * Hh * LK, 256, 0, stream>>>(s2f, a2b, NPS, 0);
  gemm(2, a2b, NPS, (long)LK * NPS, v2Th, NPS, (long)DHd * NPS, nullptr, nullptr,
       sah, DHd, (long)LK * DHd, LK, DHd, NPS, Bb * Hh, 0.f, LK, LK, 0);
  merge_heads_kernel<<<g1d(hc2), 256, 0, stream>>>(sah, sab, LK);
  // in-place residual update of the s-rows of x2
  gemm(0, sab, Dd, 0, WtpT, Dd, 0, btp, x2, x2, Dd, 0,
       Bb * LK, Dd, Dd, 1, 0.f, LK, NF, NT);

  // ---- stage 3: MLP with residual, final x written straight to d_out ----
  ln_bf16_kernel<<<Bb * NF, 256, 0, stream>>>(x2, g2c, b2c, h2b, NF, NF, 0);
  gemm(3, h2b, Dd, 0, W1T, Dd, 0, b1m, nullptr, m1, 4 * Dd, 0,
       Bb * NF, 4 * Dd, Dd, 1, 0.f, Bb * NF, Bb * NF, 0);
  gemm(0, m1, 4 * Dd, 0, W2T, 4 * Dd, 0, b2m, x2, out_x, Dd, 0,
       Bb * NF, Dd, 4 * Dd, 1, 0.f, Bb * NF, Bb * NF, 0);

  (void)in_sizes; (void)n_in; (void)out_size; (void)ws_size;
}
